// StyledMLP_55516747268437
// MI455X (gfx1250) — matrix operand order repaired and checked by reading, not compiled, not run on hardware
//
#include <hip/hip_runtime.h>

typedef float v8f  __attribute__((ext_vector_type(8)));
typedef __bf16 v16bf __attribute__((ext_vector_type(16)));
typedef unsigned int u32x4 __attribute__((ext_vector_type(4)));
typedef int v4i __attribute__((ext_vector_type(4)));

#define BATCH 16
#define CH    256      // Cin = Chid = Cout
#define NCOL  16384
#define SDIM  512

#define INV_SQRT_S   0.044194173824159216f   // 1/sqrt(512)
#define INV_SQRT_C   0.0625f                 // 1/sqrt(256)
#define SQRT2        1.4142135623730951f
#define LSLOPE       0.2f

#define LDSK  264    // ushort stride per staged column (256 + 8 pad) = 528 B

// ---------------------------------------------------------------------------
// CDNA5 async memory->LDS path (ASYNCcnt), with synchronous fallback
// ---------------------------------------------------------------------------
#if defined(__gfx1250__) && __has_builtin(__builtin_amdgcn_global_load_async_to_lds_b128)
#define USE_ASYNC_LDS 1
#else
#define USE_ASYNC_LDS 0
#endif

#if USE_ASYNC_LDS
__device__ __forceinline__ void async_cp16(const unsigned short* g, unsigned short* l) {
  __builtin_amdgcn_global_load_async_to_lds_b128(
      (__attribute__((address_space(1))) v4i*)g,
      (__attribute__((address_space(3))) v4i*)l, 0, 0);
}
__device__ __forceinline__ void wait_async0() {
#if __has_builtin(__builtin_amdgcn_s_wait_asynccnt)
  __builtin_amdgcn_s_wait_asynccnt(0);
#else
  asm volatile("s_wait_asynccnt 0x0" ::: "memory");
#endif
}
#endif

// ---------------------------------------------------------------------------
// fp32 -> bf16 hi/lo split (round-to-nearest-even both halves)
// ---------------------------------------------------------------------------
__device__ __forceinline__ void bf16split(float v, unsigned short& h, unsigned short& l) {
  unsigned int u = __builtin_bit_cast(unsigned int, v);
  unsigned int r = u + 0x7FFFu + ((u >> 16) & 1u);
  unsigned short hs = (unsigned short)(r >> 16);
  float fh = __builtin_bit_cast(float, (unsigned int)hs << 16);
  float d = v - fh;
  unsigned int ud = __builtin_bit_cast(unsigned int, d);
  unsigned int rd = ud + 0x7FFFu + ((ud >> 16) & 1u);
  h = hs;
  l = (unsigned short)(rd >> 16);
}

__device__ __forceinline__ u32x4 pack8(const unsigned short* s) {
  u32x4 p;
  p.x = (unsigned int)s[0] | ((unsigned int)s[1] << 16);
  p.y = (unsigned int)s[2] | ((unsigned int)s[3] << 16);
  p.z = (unsigned int)s[4] | ((unsigned int)s[5] << 16);
  p.w = (unsigned int)s[6] | ((unsigned int)s[7] << 16);
  return p;
}

union FragBF {
  v16bf v;
  u32x4 u[2];
};

// ---------------------------------------------------------------------------
// style[b][ci] = dot(z[b], mw[ci]) / sqrt(S) + mb[ci]
// ---------------------------------------------------------------------------
__global__ __launch_bounds__(256) void style_kernel(
    const float* __restrict__ z, const float* __restrict__ mw,
    const float* __restrict__ mb, float* __restrict__ style) {
  const int b  = blockIdx.x;
  const int ci = threadIdx.x;
  const float* zr  = z  + (size_t)b  * SDIM;
  const float* mwr = mw + (size_t)ci * SDIM;
  float s = 0.f;
#pragma unroll 8
  for (int j = 0; j < SDIM; ++j) s += zr[j] * mwr[j];
  style[b * CH + ci] = s * INV_SQRT_S + mb[ci];
}

// ---------------------------------------------------------------------------
// Modulate + demodulate, emit split-bf16 weight planes [b][co][k]
// ---------------------------------------------------------------------------
__global__ __launch_bounds__(256) void modw_kernel(
    const float* __restrict__ w, const float* __restrict__ style,
    unsigned short* __restrict__ whi, unsigned short* __restrict__ wlo) {
  const int bco = blockIdx.x;
  const int b  = bco >> 8;
  const int co = bco & 255;
  const int ci = threadIdx.x;

  float v = w[co * CH + ci] * style[b * CH + ci] * INV_SQRT_C;

  __shared__ float red[256];
  red[ci] = v * v;
  __syncthreads();
#pragma unroll
  for (int off = 128; off > 0; off >>= 1) {
    if (ci < off) red[ci] += red[ci + off];
    __syncthreads();
  }
  const float demod = rsqrtf(red[0] + 1e-8f);
  unsigned short h, l;
  bf16split(v * demod, h, l);
  const size_t idx = ((size_t)b * CH + co) * CH + ci;
  whi[idx] = h;
  wlo[idx] = l;
}

// ---------------------------------------------------------------------------
// Transpose + split layer-0 input x (B, CH, N) fp32 -> (B, N, CH) bf16 hi/lo
// ---------------------------------------------------------------------------
__global__ __launch_bounds__(256) void split_x_kernel(
    const float* __restrict__ X, unsigned short* __restrict__ Ph,
    unsigned short* __restrict__ Pl) {
  const int b = blockIdx.y, col0 = blockIdx.x * 64, t = threadIdx.x;
  __shared__ float tile[64][65];
  const int cin = t & 63, kr = t >> 6;   // load indexing
  const int c2  = t >> 2, q  = t & 3;    // store indexing

  for (int kc = 0; kc < CH; kc += 64) {
    __syncthreads();
#pragma unroll
    for (int i = 0; i < 16; ++i) {
      const int k = kr + 4 * i;
      tile[k][cin] = X[((size_t)b * CH + kc + k) * NCOL + col0 + cin];
    }
    __syncthreads();
    unsigned short hs[16], ls[16];
#pragma unroll
    for (int j = 0; j < 16; ++j) bf16split(tile[q * 16 + j][c2], hs[j], ls[j]);
    const size_t base = ((size_t)b * NCOL + col0 + c2) * CH + kc + q * 16;
    *(u32x4*)(Ph + base)     = pack8(hs);
    *(u32x4*)(Ph + base + 8) = pack8(hs + 8);
    *(u32x4*)(Pl + base)     = pack8(ls);
    *(u32x4*)(Pl + base + 8) = pack8(ls + 8);
  }
}

// ---------------------------------------------------------------------------
// Out[b] = act( W[b](256x256) @ X[b](256xN) + bias )
// Split-bf16 GEMM: 3x V_WMMA_F32_16X16X32_BF16 per 16x16x32 tile.
// Workgroup: 8 waves, owns (b, 64 cols) x all 256 out rows; X staged in LDS
// via GLOBAL_LOAD_ASYNC_TO_LDS_B128 (ASYNCcnt) when available.
// write_f32 == 0: emit split-bf16 (B,N,CH) planes; == 1: fp32 (B,CH,N).
// ---------------------------------------------------------------------------
__global__ __launch_bounds__(256) void gemm_bf16x3_kernel(
    const unsigned short* __restrict__ Xhi, const unsigned short* __restrict__ Xlo,
    const unsigned short* __restrict__ Whi, const unsigned short* __restrict__ Wlo,
    const float* __restrict__ bias,
    float* __restrict__ OutF,
    unsigned short* __restrict__ Ohi, unsigned short* __restrict__ Olo,
    const int write_f32) {
  const int b    = blockIdx.y;
  const int col0 = blockIdx.x * 64;
  const int t    = threadIdx.x;
  const int wv   = t >> 5;
  const int lane = t & 31;
  const int lhalf = lane >> 4;
  const int lmod  = lane & 15;

  __shared__ unsigned short xsh[64 * LDSK];  // 33 KB hi plane (padded columns)
  __shared__ unsigned short xsl[64 * LDSK];  // 33 KB lo plane
  __shared__ float bsh[CH];

  bsh[t] = bias[t];

  // ---- stage full-K activation slab for 64 columns ----
  {
    const int c = t >> 2;
    const int q = t & 3;
    const size_t gbase = ((size_t)b * NCOL + col0 + c) * CH + q * 64;
    const unsigned short* gh = Xhi + gbase;
    const unsigned short* gl = Xlo + gbase;
    unsigned short* sh = xsh + c * LDSK + q * 64;
    unsigned short* sl = xsl + c * LDSK + q * 64;
#if USE_ASYNC_LDS
#pragma unroll
    for (int i = 0; i < 8; ++i) {
      async_cp16(gh + i * 8, sh + i * 8);
      async_cp16(gl + i * 8, sl + i * 8);
    }
    wait_async0();
#else
#pragma unroll
    for (int i = 0; i < 8; ++i) {
      *(u32x4*)(sh + i * 8) = *(const u32x4*)(gh + i * 8);
      *(u32x4*)(sl + i * 8) = *(const u32x4*)(gl + i * 8);
    }
#endif
  }
  __syncthreads();

  // A-fragment row pointers: m = 32*wv + 16*rt + lmod, K-major in memory
  const size_t wrow = ((size_t)b * CH + 32 * wv + lmod) * CH;
  const unsigned short* wh0 = Whi + wrow;
  const unsigned short* wl0 = Wlo + wrow;

  v8f acc[2][4];
#pragma unroll
  for (int rt = 0; rt < 2; ++rt)
#pragma unroll
    for (int ct = 0; ct < 4; ++ct)
      acc[rt][ct] = (v8f)(0.f);

  for (int kk = 0; kk < CH; kk += 32) {
    // A fragments (16-bit A 16x32 layout): lane<16 K {kk..kk+7, kk+16..kk+23},
    // lane>=16 K {kk+8..kk+15, kk+24..kk+31}
    const int koff = kk + 8 * lhalf;
    FragBF ah[2], al[2];
#pragma unroll
    for (int rt = 0; rt < 2; ++rt) {
      const unsigned short* ph = wh0 + rt * 16 * CH + koff;
      const unsigned short* pl = wl0 + rt * 16 * CH + koff;
      ah[rt].u[0] = *(const u32x4*)(ph);
      ah[rt].u[1] = *(const u32x4*)(ph + 16);
      al[rt].u[0] = *(const u32x4*)(pl);
      al[rt].u[1] = *(const u32x4*)(pl + 16);
    }
    // B fragments (16-bit B 32x16 layout): lane<16 K {kk..kk+7, kk+16..kk+23},
    // lane>=16 K {kk+8..kk+15, kk+24..kk+31}
    const int kbB = kk + 8 * lhalf;
#pragma unroll
    for (int ct = 0; ct < 4; ++ct) {
      const int c = 16 * ct + lmod;
      const unsigned short* sh = xsh + c * LDSK + kbB;
      const unsigned short* sl = xsl + c * LDSK + kbB;
      FragBF bh, bl;
      bh.u[0] = *(const u32x4*)(sh);
      bh.u[1] = *(const u32x4*)(sh + 16);
      bl.u[0] = *(const u32x4*)(sl);
      bl.u[1] = *(const u32x4*)(sl + 16);
#pragma unroll
      for (int rt = 0; rt < 2; ++rt) {
        acc[rt][ct] = __builtin_amdgcn_wmma_f32_16x16x32_bf16(
            false, ah[rt].v, false, bh.v, (short)0, acc[rt][ct], false, false);
        acc[rt][ct] = __builtin_amdgcn_wmma_f32_16x16x32_bf16(
            false, ah[rt].v, false, bl.v, (short)0, acc[rt][ct], false, false);
        acc[rt][ct] = __builtin_amdgcn_wmma_f32_16x16x32_bf16(
            false, al[rt].v, false, bh.v, (short)0, acc[rt][ct], false, false);
      }
    }
  }

  // ---- epilogue: bias + leaky_relu * sqrt(2) ----
#pragma unroll
  for (int rt = 0; rt < 2; ++rt) {
    const int rowbase = 32 * wv + 16 * rt + 8 * lhalf;
#pragma unroll
    for (int ct = 0; ct < 4; ++ct) {
      const int col = col0 + 16 * ct + lmod;
      const v8f a = acc[rt][ct];
      float vals[8];
#pragma unroll
      for (int r = 0; r < 8; ++r) {
        float v = a[r] + bsh[rowbase + r];
        vals[r] = (v >= 0.f ? v : LSLOPE * v) * SQRT2;
      }
      if (write_f32) {
        float* op = OutF + ((size_t)b * CH + rowbase) * NCOL + col;
#pragma unroll
        for (int r = 0; r < 8; ++r) op[(size_t)r * NCOL] = vals[r];
      } else {
        unsigned short hs[8], ls[8];
#pragma unroll
        for (int r = 0; r < 8; ++r) bf16split(vals[r], hs[r], ls[r]);
        const size_t base = ((size_t)b * NCOL + col) * CH + rowbase;
        *(u32x4*)(Ohi + base) = pack8(hs);
        *(u32x4*)(Olo + base) = pack8(ls);
      }
    }
  }
}

// ---------------------------------------------------------------------------
extern "C" void kernel_launch(void* const* d_in, const int* in_sizes, int n_in,
                              void* d_out, int out_size, void* d_ws, size_t ws_size,
                              hipStream_t stream) {
  const float* x = (const float*)d_in[0];
  const float* z = (const float*)d_in[1];
  const float* w[3]  = {(const float*)d_in[2], (const float*)d_in[6],  (const float*)d_in[10]};
  const float* mw[3] = {(const float*)d_in[3], (const float*)d_in[7],  (const float*)d_in[11]};
  const float* mb[3] = {(const float*)d_in[4], (const float*)d_in[8],  (const float*)d_in[12]};
  const float* bs[3] = {(const float*)d_in[5], (const float*)d_in[9],  (const float*)d_in[13]};
  float* out = (float*)d_out;

  char* p = (char*)d_ws;
  float* style = (float*)p;          p += (size_t)4096 * 4;
  unsigned short* whi = (unsigned short*)p;  p += (size_t)BATCH * CH * CH * 2;
  unsigned short* wlo = (unsigned short*)p;  p += (size_t)BATCH * CH * CH * 2;
  unsigned short* acth = (unsigned short*)p; p += (size_t)BATCH * NCOL * CH * 2;
  unsigned short* actl = (unsigned short*)p;

  const dim3 grid2(NCOL / 64, BATCH);

  // one-time: transpose+split layer-0 input into activation planes
  split_x_kernel<<<grid2, 256, 0, stream>>>(x, acth, actl);

  for (int l = 0; l < 3; ++l) {
    style_kernel<<<BATCH, 256, 0, stream>>>(z, mw[l], mb[l], style);
    modw_kernel<<<BATCH * CH, 256, 0, stream>>>(w[l], style, whi, wlo);
    // layers 0,1: in-place split-bf16 activation planes; layer 2: fp32 d_out
    gemm_bf16x3_kernel<<<grid2, 256, 0, stream>>>(
        acth, actl, whi, wlo, bs[l], out, acth, actl, (l == 2) ? 1 : 0);
  }
}
